// GQAttention_80384607912347
// MI455X (gfx1250) — matrix operand order repaired and checked by reading, not compiled, not run on hardware
//
#include <hip/hip_runtime.h>

typedef __attribute__((ext_vector_type(16))) __bf16 v16bf;
typedef __attribute__((ext_vector_type(8)))  __bf16 v8bf;
typedef __attribute__((ext_vector_type(8)))  float  v8f;
typedef __attribute__((ext_vector_type(4)))  float  v4f;
typedef __attribute__((ext_vector_type(4)))  int    v4i;
typedef __attribute__((ext_vector_type(4)))  unsigned v4u;

#define B_    2
#define H_    16
#define HKV_  4
#define G_    4
#define SQ_   2048
#define SKV_  2048
#define D_    64
#define NCHUNK_ (SKV_ / 32)

// native bf16 convert (gfx1250 has hardware packed bf16 cvt)
static __device__ __forceinline__ __bf16 f2bf(float f) { return (__bf16)f; }

static __device__ __forceinline__ unsigned pack2bf(float lo, float hi) {
    unsigned short a = __builtin_bit_cast(unsigned short, (__bf16)lo);
    unsigned short b = __builtin_bit_cast(unsigned short, (__bf16)hi);
    return (unsigned)a | ((unsigned)b << 16);
}

// ---------------------------------------------------------------------------
// Prep kernels: one-time K/V bf16 conversion into workspace.
//   kbf: [B][HKV][SKV][D]   row-major bf16
//   vbf: [B][HKV][NCHUNK][64][32] bf16  (V chunk transposed: [d][kv])
// ---------------------------------------------------------------------------
__global__ __launch_bounds__(256)
void prep_k_kernel(const float* __restrict__ k, __bf16* __restrict__ kbf)
{
    size_t tid = (size_t)blockIdx.x * 256 + threadIdx.x;   // N/8 threads
    const float* src = k + tid * 8;
    v4f a = *(const v4f*)(src);
    v4f b = *(const v4f*)(src + 4);
    v4u pk;
    pk[0] = pack2bf(a[0], a[1]);
    pk[1] = pack2bf(a[2], a[3]);
    pk[2] = pack2bf(b[0], b[1]);
    pk[3] = pack2bf(b[2], b[3]);
    ((v4u*)kbf)[tid] = pk;
}

__global__ __launch_bounds__(256)
void prep_v_kernel(const float* __restrict__ v, __bf16* __restrict__ vbf)
{
    // one block per (b,hkv,chunk) tile
    const int bc    = blockIdx.x;            // 0 .. B*HKV*NCHUNK-1
    const int chunk = bc & (NCHUNK_ - 1);
    const int bh    = bc >> 6;               // b*HKV + hkv   (NCHUNK==64)
    const int pp = threadIdx.x & 15;         // kv pair: kv = 2pp, 2pp+1
    const int dg = threadIdx.x >> 4;         // 0..15 -> d = dg*4 .. dg*4+3

    const float* r0 = v + ((size_t)bh * SKV_ + chunk * 32 + 2 * pp) * D_ + dg * 4;
    const float* r1 = r0 + D_;
    v4f x0 = *(const v4f*)r0;
    v4f x1 = *(const v4f*)r1;

    unsigned* dst = (unsigned*)(vbf + (size_t)bc * 2048);   // [64][32] bf16 tile
#pragma unroll
    for (int j = 0; j < 4; ++j)
        dst[(dg * 4 + j) * 16 + pp] = pack2bf(x0[j], x1[j]);
}

// ---------------------------------------------------------------------------
// Main flash-attention kernel (bf16 K/V from workspace, no barriers).
// Each wave owns a 16-row Q tile; 8 independent waves per block.
// ---------------------------------------------------------------------------
__global__ __launch_bounds__(256)
void gqa_flash_wmma_kernel(const float* __restrict__ q,
                           const __bf16* __restrict__ kbf,
                           const __bf16* __restrict__ vbf,
                           const int*   __restrict__ qseg,
                           const int*   __restrict__ kvseg,
                           float*       __restrict__ out)
{
    const int lane = threadIdx.x & 31;
    const int wave = threadIdx.x >> 5;
    const int n    = lane & 15;   // this lane's Q-row within the wave tile
    const int hig  = lane >> 4;   // 0: lanes 0-15, 1: lanes 16-31

    const int h  = blockIdx.y;
    const int b  = blockIdx.z;
    const int bh = b * HKV_ + (h / G_);
    const int qbase = (blockIdx.x * 8 + wave) * 16;

    __shared__ float sA[8][16];   // per-wave row-broadcast scratch

    const int* qsb  = qseg  + b * SQ_;
    const int* kvsb = kvseg + b * SKV_;

    // sorted ids -> this wave's q segment range = endpoints
    const int qmn = qsb[qbase];
    const int qmx = qsb[qbase + 15];
    const int rowseg = qsb[qbase + n];

    // ---- Q as B-fragment (32d x 16q), two d-chunks; softmax scale folded ----
    const float qscale = 0.125f * 1.44269504088896340736f; // 1/sqrt(D)*log2(e)
    v16bf Qb[2];
    {
        const float* qrow = q + (((size_t)b * H_ + h) * SQ_ + qbase + n) * D_;
#pragma unroll
        for (int c = 0; c < 2; ++c) {
            const float* p = qrow + c * 32 + hig * 16;  // 64B contiguous
#pragma unroll
            for (int j = 0; j < 16; ++j) Qb[c][j] = f2bf(p[j] * qscale);
        }
    }

    v8f O[4];
#pragma unroll
    for (int dt = 0; dt < 4; ++dt) O[dt] = (v8f){};
    float m = -1e30f, l = 0.0f;

    const __bf16* kb = kbf + (size_t)bh * SKV_ * D_;
    const __bf16* vt = vbf + (size_t)bh * NCHUNK_ * 2048;

    for (int kv0 = 0; kv0 < SKV_; kv0 += 32) {
        // ---- uniform chunk skip: sorted ids -> endpoints are min/max ----
        int kmn = kvsb[kv0];
        int kmx = kvsb[kv0 + 31];
        int skip = __builtin_amdgcn_readfirstlane((int)((kmx < qmn) | (kmn > qmx)));
        if (skip) continue;

        // speculative prefetch of next chunk's bf16 K rows / V tile
        {
            int kvp = (kv0 + 32) & (SKV_ - 1);
            __builtin_prefetch(kb + (size_t)(kvp + lane) * D_, 0, 1);
            __builtin_prefetch(vt + (size_t)(kvp >> 5) * 2048 + lane * 64, 0, 1);
        }

        // ---- S^T = K * Q^T : two 16(kv) x 16(q) tiles ----
        v8f St[2];
#pragma unroll
        for (int t = 0; t < 2; ++t) {
            v8f acc = (v8f){};
#pragma unroll
            for (int c = 0; c < 2; ++c) {
                // A fragment: lane holds K-row (kv0 + t*16 + n),
                // d entries c*32 + hig*16+0..15
                const __bf16* kp = kb + (size_t)(kv0 + t * 16 + n) * D_
                                 + c * 32 + hig * 16;
                v8bf lo = *(const v8bf*)kp;
                v8bf hi = *(const v8bf*)(kp + 8);
                v16bf Ka;
#pragma unroll
                for (int j = 0; j < 8; ++j) { Ka[j] = lo[j]; Ka[8 + j] = hi[j]; }
                acc = __builtin_amdgcn_wmma_f32_16x16x32_bf16(
                        false, Ka, false, Qb[c], (short)0, acc, false, false);
            }
            St[t] = acc;
        }

        // ---- segment mask: this lane's kv rows (vector int loads) ----
        v4i ks0a = *(const v4i*)(kvsb + kv0 + hig * 8);
        v4i ks0b = *(const v4i*)(kvsb + kv0 + hig * 8 + 4);
        v4i ks1a = *(const v4i*)(kvsb + kv0 + 16 + hig * 8);
        v4i ks1b = *(const v4i*)(kvsb + kv0 + 16 + hig * 8 + 4);

        float sm[16];
#pragma unroll
        for (int r = 0; r < 4; ++r) {
            sm[r]      = (ks0a[r] == rowseg) ? St[0][r]     : -1e30f;
            sm[4 + r]  = (ks0b[r] == rowseg) ? St[0][4 + r] : -1e30f;
            sm[8 + r]  = (ks1a[r] == rowseg) ? St[1][r]     : -1e30f;
            sm[12 + r] = (ks1b[r] == rowseg) ? St[1][4 + r] : -1e30f;
        }

        // ---- online softmax: in-lane reduce over 16 + one xor-16 shuffle ----
        float c01 = fmaxf(sm[0], sm[1]),   c23 = fmaxf(sm[2], sm[3]);
        float c45 = fmaxf(sm[4], sm[5]),   c67 = fmaxf(sm[6], sm[7]);
        float c89 = fmaxf(sm[8], sm[9]),   cab = fmaxf(sm[10], sm[11]);
        float ccd = fmaxf(sm[12], sm[13]), cef = fmaxf(sm[14], sm[15]);
        float cm = fmaxf(fmaxf(fmaxf(c01, c23), fmaxf(c45, c67)),
                         fmaxf(fmaxf(c89, cab), fmaxf(ccd, cef)));
        cm = fmaxf(cm, __shfl_xor(cm, 16, 32));
        float mn = fmaxf(m, cm);

        // ---- O-rescale only when some row's max actually moved ----
        if (__any(mn > m)) {
            float alpha = exp2f(m - mn);
            l *= alpha;
            sA[wave][n] = alpha;        // lanes n and n+16 write same value
            v4f a0 = *(const v4f*)&sA[wave][hig * 8];
            v4f a1 = *(const v4f*)&sA[wave][hig * 8 + 4];
#pragma unroll
            for (int dt = 0; dt < 4; ++dt) {
#pragma unroll
                for (int r = 0; r < 4; ++r) {
                    O[dt][r]     *= a0[r];
                    O[dt][4 + r] *= a1[r];
                }
            }
        }

        float p[16];
#pragma unroll
        for (int j = 0; j < 16; ++j)
            p[j] = (sm[j] > -5e29f) ? exp2f(sm[j] - mn) : 0.0f;
        float t0 = (p[0] + p[1]) + (p[2] + p[3]);
        float t1 = (p[4] + p[5]) + (p[6] + p[7]);
        float t2 = (p[8] + p[9]) + (p[10] + p[11]);
        float t3 = (p[12] + p[13]) + (p[14] + p[15]);
        float ps = (t0 + t1) + (t2 + t3);
        ps += __shfl_xor(ps, 16, 32);
        l += ps;
        m = mn;

        // ---- P A-fragment: registers already match A layout, just convert ----
        v16bf Pa;
#pragma unroll
        for (int j = 0; j < 8; ++j) {
            Pa[j]     = f2bf(p[j]);        // tile0 -> K 0..7 / 8..15
            Pa[8 + j] = f2bf(p[8 + j]);    // tile1 -> K 16..23 / 24..31
        }

        // ---- O += P * V : V B-fragments straight from pre-packed global ----
        const __bf16* vtile = vt + (size_t)(kv0 >> 5) * 2048;
#pragma unroll
        for (int dt = 0; dt < 4; ++dt) {
            const __bf16* vrow = vtile + (dt * 16 + n) * 32 + hig * 8;
            v8bf w0 = *(const v8bf*)vrow;
            v8bf w1 = *(const v8bf*)(vrow + 16);
            v16bf Vb;
#pragma unroll
            for (int j = 0; j < 8; ++j) { Vb[j] = w0[j]; Vb[8 + j] = w1[j]; }
            O[dt] = __builtin_amdgcn_wmma_f32_16x16x32_bf16(
                      false, Pa, false, Vb, (short)0, O[dt], false, false);
        }
    }

    // ---- epilogue: out = O / l (0 for fully-masked rows) ----
    float linv = (l > 0.0f) ? (1.0f / l) : 0.0f;
    sA[wave][n] = linv;                  // same-wave DS ordering guarantees RAW
    v4f l0 = *(const v4f*)&sA[wave][hig * 8];
    v4f l1 = *(const v4f*)&sA[wave][hig * 8 + 4];

    float* ob = out + (((size_t)b * H_ + h) * SQ_ + qbase) * D_;
#pragma unroll
    for (int r = 0; r < 4; ++r) {
#pragma unroll
        for (int dt = 0; dt < 4; ++dt) {
            ob[(size_t)(hig * 8 + r) * D_ + dt * 16 + n]     = O[dt][r]     * l0[r];
            ob[(size_t)(hig * 8 + 4 + r) * D_ + dt * 16 + n] = O[dt][4 + r] * l1[r];
        }
    }
}

// ---------------------------------------------------------------------------
// Fallback (self-contained, converts in-kernel) if workspace is too small.
// ---------------------------------------------------------------------------
__global__ __launch_bounds__(128)
void gqa_flash_wmma_fb_kernel(const float* __restrict__ q,
                              const float* __restrict__ k,
                              const float* __restrict__ v,
                              const int*   __restrict__ qseg,
                              const int*   __restrict__ kvseg,
                              float*       __restrict__ out)
{
    const int lane = threadIdx.x & 31;
    const int wave = threadIdx.x >> 5;
    const int n    = lane & 15;
    const int hig  = lane >> 4;

    const int qblk = blockIdx.x;
    const int h    = blockIdx.y;
    const int b    = blockIdx.z;
    const int hkv  = h / G_;
    const int qbase_blk = qblk * 64;
    const int qbase     = qbase_blk + wave * 16;

    __shared__ __bf16 sV[64][32];
    __shared__ float  sA[4][16];

    const int* qsb  = qseg  + b * SQ_;
    const int* kvsb = kvseg + b * SKV_;
    const int qmn = qsb[qbase_blk];
    const int qmx = qsb[qbase_blk + 63];
    const int rowseg = qsb[qbase + n];

    const float qscale = 0.125f * 1.44269504088896340736f;
    v16bf Qb[2];
    {
        const float* qrow = q + (((size_t)b * H_ + h) * SQ_ + qbase + n) * D_;
#pragma unroll
        for (int c = 0; c < 2; ++c) {
            const float* p = qrow + c * 32 + hig * 16;
#pragma unroll
            for (int j = 0; j < 16; ++j) Qb[c][j] = f2bf(p[j] * qscale);
        }
    }

    v8f O[4];
#pragma unroll
    for (int dt = 0; dt < 4; ++dt) O[dt] = (v8f){};
    float m = -1e30f, l = 0.0f;

    const float* kb = k + (size_t)(b * HKV_ + hkv) * SKV_ * D_;
    const float* vb = v + (size_t)(b * HKV_ + hkv) * SKV_ * D_;

    for (int kv0 = 0; kv0 < SKV_; kv0 += 32) {
        int kmn = kvsb[kv0];
        int kmx = kvsb[kv0 + 31];
        int skip = __builtin_amdgcn_readfirstlane((int)((kmx < qmn) | (kmn > qmx)));
        if (skip) continue;

        __syncthreads();
        {
            int pp = threadIdx.x & 15;
            int dg = threadIdx.x >> 4;
            const float* v0p = vb + (size_t)(kv0 + 2 * pp) * D_ + dg * 8;
            const float* v1p = v0p + D_;
            v4f x0a = *(const v4f*)(v0p);
            v4f x0b = *(const v4f*)(v0p + 4);
            v4f x1a = *(const v4f*)(v1p);
            v4f x1b = *(const v4f*)(v1p + 4);
#pragma unroll
            for (int j = 0; j < 4; ++j)
                ((unsigned*)&sV[dg * 8 + j][0])[pp] = pack2bf(x0a[j], x1a[j]);
#pragma unroll
            for (int j = 0; j < 4; ++j)
                ((unsigned*)&sV[dg * 8 + 4 + j][0])[pp] = pack2bf(x0b[j], x1b[j]);
        }
        __syncthreads();

        v8f St[2];
#pragma unroll
        for (int t = 0; t < 2; ++t) {
            v8f acc = (v8f){};
#pragma unroll
            for (int c = 0; c < 2; ++c) {
                const float* kp = kb + (size_t)(kv0 + t * 16 + n) * D_
                                + c * 32 + hig * 16;
                v16bf Ka;
#pragma unroll
                for (int j = 0; j < 8; ++j) {
                    Ka[j]     = f2bf(kp[j]);
                    Ka[8 + j] = f2bf(kp[8 + j]);
                }
                acc = __builtin_amdgcn_wmma_f32_16x16x32_bf16(
                        false, Ka, false, Qb[c], (short)0, acc, false, false);
            }
            St[t] = acc;
        }

        v4i ks0a = *(const v4i*)(kvsb + kv0 + hig * 8);
        v4i ks0b = *(const v4i*)(kvsb + kv0 + hig * 8 + 4);
        v4i ks1a = *(const v4i*)(kvsb + kv0 + 16 + hig * 8);
        v4i ks1b = *(const v4i*)(kvsb + kv0 + 16 + hig * 8 + 4);

        float sm[16];
#pragma unroll
        for (int r = 0; r < 4; ++r) {
            sm[r]      = (ks0a[r] == rowseg) ? St[0][r]     : -1e30f;
            sm[4 + r]  = (ks0b[r] == rowseg) ? St[0][4 + r] : -1e30f;
            sm[8 + r]  = (ks1a[r] == rowseg) ? St[1][r]     : -1e30f;
            sm[12 + r] = (ks1b[r] == rowseg) ? St[1][4 + r] : -1e30f;
        }

        float c01 = fmaxf(sm[0], sm[1]),   c23 = fmaxf(sm[2], sm[3]);
        float c45 = fmaxf(sm[4], sm[5]),   c67 = fmaxf(sm[6], sm[7]);
        float c89 = fmaxf(sm[8], sm[9]),   cab = fmaxf(sm[10], sm[11]);
        float ccd = fmaxf(sm[12], sm[13]), cef = fmaxf(sm[14], sm[15]);
        float cm = fmaxf(fmaxf(fmaxf(c01, c23), fmaxf(c45, c67)),
                         fmaxf(fmaxf(c89, cab), fmaxf(ccd, cef)));
        cm = fmaxf(cm, __shfl_xor(cm, 16, 32));
        float mn    = fmaxf(m, cm);
        float alpha = exp2f(m - mn);

        float p[16];
#pragma unroll
        for (int j = 0; j < 16; ++j)
            p[j] = (sm[j] > -5e29f) ? exp2f(sm[j] - mn) : 0.0f;
        float t0 = (p[0] + p[1]) + (p[2] + p[3]);
        float t1 = (p[4] + p[5]) + (p[6] + p[7]);
        float t2 = (p[8] + p[9]) + (p[10] + p[11]);
        float t3 = (p[12] + p[13]) + (p[14] + p[15]);
        float ps = (t0 + t1) + (t2 + t3);
        ps += __shfl_xor(ps, 16, 32);
        l = l * alpha + ps;
        m = mn;

        sA[wave][n] = alpha;
        v4f a0 = *(const v4f*)&sA[wave][hig * 8];
        v4f a1 = *(const v4f*)&sA[wave][hig * 8 + 4];
#pragma unroll
        for (int dt = 0; dt < 4; ++dt) {
#pragma unroll
            for (int r = 0; r < 4; ++r) {
                O[dt][r]     *= a0[r];
                O[dt][4 + r] *= a1[r];
            }
        }

        v16bf Pa;
#pragma unroll
        for (int j = 0; j < 8; ++j) {
            Pa[j]     = f2bf(p[j]);
            Pa[8 + j] = f2bf(p[8 + j]);
        }

#pragma unroll
        for (int dt = 0; dt < 4; ++dt) {
            const v8bf* b0 = (const v8bf*)&sV[dt * 16 + n][hig * 8];
            const v8bf* b1 = (const v8bf*)&sV[dt * 16 + n][16 + hig * 8];
            v8bf w0 = *b0, w1 = *b1;
            v16bf Vb;
#pragma unroll
            for (int j = 0; j < 8; ++j) { Vb[j] = w0[j]; Vb[8 + j] = w1[j]; }
            O[dt] = __builtin_amdgcn_wmma_f32_16x16x32_bf16(
                      false, Pa, false, Vb, (short)0, O[dt], false, false);
        }
    }

    float linv = (l > 0.0f) ? (1.0f / l) : 0.0f;
    sA[wave][n] = linv;
    v4f l0 = *(const v4f*)&sA[wave][hig * 8];
    v4f l1 = *(const v4f*)&sA[wave][hig * 8 + 4];

    float* ob = out + (((size_t)b * H_ + h) * SQ_ + qbase) * D_;
#pragma unroll
    for (int r = 0; r < 4; ++r) {
#pragma unroll
        for (int dt = 0; dt < 4; ++dt) {
            ob[(size_t)(hig * 8 + r) * D_ + dt * 16 + n]     = O[dt][r]     * l0[r];
            ob[(size_t)(hig * 8 + 4 + r) * D_ + dt * 16 + n] = O[dt][4 + r] * l1[r];
        }
    }
}

extern "C" void kernel_launch(void* const* d_in, const int* in_sizes, int n_in,
                              void* d_out, int out_size, void* d_ws, size_t ws_size,
                              hipStream_t stream) {
    (void)in_sizes; (void)n_in; (void)out_size;
    const float* q     = (const float*)d_in[0];
    const float* k     = (const float*)d_in[1];
    const float* v     = (const float*)d_in[2];
    const int*   qsid  = (const int*)d_in[3];
    const int*   kvsid = (const int*)d_in[4];
    float*       out   = (float*)d_out;

    const size_t kv_elems = (size_t)B_ * HKV_ * SKV_ * D_;     // 1,048,576 each
    const size_t ws_need  = kv_elems * 2 * sizeof(__bf16);     // K + V bf16

    if (ws_size >= ws_need) {
        __bf16* kbf = (__bf16*)d_ws;
        __bf16* vbf = kbf + kv_elems;

        prep_k_kernel<<<dim3((unsigned)(kv_elems / 8 / 256)), dim3(256), 0, stream>>>(k, kbf);
        prep_v_kernel<<<dim3(B_ * HKV_ * NCHUNK_), dim3(256), 0, stream>>>(v, vbf);

        dim3 grid(SQ_ / 128, H_, B_);   // 8 waves/block, 16 q-rows/wave
        gqa_flash_wmma_kernel<<<grid, dim3(256), 0, stream>>>(q, kbf, vbf, qsid, kvsid, out);
    } else {
        dim3 grid(SQ_ / 64, H_, B_);
        gqa_flash_wmma_fb_kernel<<<grid, dim3(128), 0, stream>>>(q, k, v, qsid, kvsid, out);
    }
}
